// CascadedAttention_89627377533499
// MI455X (gfx1250) — matrix operand order repaired and checked by reading, not compiled, not run on hardware
//
#include <hip/hip_runtime.h>

// ---------------------------------------------------------------------------
// Cascaded Bahdanau-attention + GRU for MI455X (gfx1250, wave32).
//  B=256, T=150, F=1024, U=28.
//  Strategy:
//   * Convert x and Ua (transposed) to bf16 once; UaH = x@Ua + Ba2 via
//     v_wmma_f32_16x16x32_bf16, stored bf16. x_bf16 + UaH_bf16 = 157MB, fits
//     the 192MB L2, so the 150 recurrent steps stream from L2 not HBM.
//   * One fused kernel per step: block b owns batch row b (256 thr = 8 waves).
//   * Ba3 omitted: softmax is shift-invariant.
// Workspace: 78,643,200 (x_bf16) + 2,097,152 (UaT_bf16) + 78,643,200 (UaH_bf16)
//            + 28,672 (h)  ~= 160MB.
// ---------------------------------------------------------------------------

#define B_ 256
#define T_ 150
#define F_ 1024
#define U_ 28
#define G3_ 84  // 3*U

typedef __attribute__((ext_vector_type(16))) __bf16 v16bf;
typedef __attribute__((ext_vector_type(8)))  float  v8f;
typedef __attribute__((ext_vector_type(8)))  __bf16 bf8v;
typedef __attribute__((ext_vector_type(4)))  __bf16 bf4v;

struct Q2 { uint4 lo, hi; };

__device__ __forceinline__ float fast_tanh(float x) {
  x = fminf(10.f, fmaxf(-10.f, x));
  float e = __expf(2.f * x);
  return (e - 1.f) / (e + 1.f);
}
__device__ __forceinline__ float fast_sigmoid(float x) {
  return 1.f / (1.f + __expf(-x));
}

// ---------------------------------------------------------------------------
// Kernel 1: f32 -> bf16 conversion of x, and transposed bf16 copy of Ua.
// ---------------------------------------------------------------------------
__global__ __launch_bounds__(256) void convert_kernel(
    const float* __restrict__ x, const float* __restrict__ Ua,
    __bf16* __restrict__ xbf, __bf16* __restrict__ uat) {
  size_t idx = (size_t)blockIdx.x * 256 + threadIdx.x;
  size_t stride = (size_t)gridDim.x * 256;
  const size_t NX = (size_t)B_ * T_ * F_;
  for (size_t i = idx; i < NX; i += stride) xbf[i] = (__bf16)x[i];
  const size_t NU = (size_t)F_ * F_;
  for (size_t i = idx; i < NU; i += stride) {
    size_t n = i >> 10, k = i & 1023;       // uat[n,k] = Ua[k,n]
    uat[i] = (__bf16)Ua[k * F_ + n];
  }
}

// ---------------------------------------------------------------------------
// Kernel 2: UaH = x @ Ua + Ba2  (bf16 WMMA, f32 accumulate, bf16 out)
//  A = xbf (M=38400, K=1024) row-major; B = uat (N=1024, K=1024) row-major
//  (i.e. Ua transposed). Block tile 128(M)x64(N); 8 waves as 4x2 grid of
//  32x32 wave tiles (2x2 WMMA 16x16 tiles each).
// ---------------------------------------------------------------------------
__global__ __launch_bounds__(256) void wmma_gemm_kernel(
    const __bf16* __restrict__ A, const __bf16* __restrict__ Bt,
    const float* __restrict__ Ba2, __bf16* __restrict__ C) {
  constexpr int K = F_, N = F_;
  const int tid  = threadIdx.x;
  const int lane = tid & 31;
  const int wave = tid >> 5;
  const int waveM = wave & 3;          // 0..3
  const int waveN = wave >> 2;         // 0..1
  const int mblk  = blockIdx.y * 128 + waveM * 32;
  const int nblk  = blockIdx.x * 64  + waveN * 32;
  const int l16   = lane & 15;
  const int khalf = lane >> 4;

  v8f acc[2][2] = {};

  const __bf16* arow0 = A  + (size_t)(mblk + l16) * K;
  const __bf16* arow1 = arow0 + (size_t)16 * K;
  const __bf16* brow0 = Bt + (size_t)(nblk + l16) * K;
  const __bf16* brow1 = brow0 + (size_t)16 * K;

  for (int kc = 0; kc < K; kc += 32) {
    // A fragment: lane = row m; VGPR0-3 hold K = khalf*8+0..7,
    // VGPR4-7 hold K = 16+khalf*8+0..7 (ISA 16-bit 16x32 A layout).
    Q2 qa0, qa1;
    qa0.lo = *(const uint4*)(arow0 + kc + khalf * 8);
    qa0.hi = *(const uint4*)(arow0 + kc + 16 + khalf * 8);
    qa1.lo = *(const uint4*)(arow1 + kc + khalf * 8);
    qa1.hi = *(const uint4*)(arow1 + kc + 16 + khalf * 8);
    v16bf a0 = __builtin_bit_cast(v16bf, qa0);
    v16bf a1 = __builtin_bit_cast(v16bf, qa1);
    // B fragment: lane = col n; same K layout as A: VGPR0-3 hold
    // K = khalf*8+0..7, VGPR4-7 hold K = 16+khalf*8+0..7.
    Q2 qb0, qb1;
    qb0.lo = *(const uint4*)(brow0 + kc + khalf * 8);
    qb0.hi = *(const uint4*)(brow0 + kc + 16 + khalf * 8);
    qb1.lo = *(const uint4*)(brow1 + kc + khalf * 8);
    qb1.hi = *(const uint4*)(brow1 + kc + 16 + khalf * 8);
    v16bf b0 = __builtin_bit_cast(v16bf, qb0);
    v16bf b1 = __builtin_bit_cast(v16bf, qb1);

    acc[0][0] = __builtin_amdgcn_wmma_f32_16x16x32_bf16(
        false, a0, false, b0, (short)0, acc[0][0], false, false);
    acc[0][1] = __builtin_amdgcn_wmma_f32_16x16x32_bf16(
        false, a0, false, b1, (short)0, acc[0][1], false, false);
    acc[1][0] = __builtin_amdgcn_wmma_f32_16x16x32_bf16(
        false, a1, false, b0, (short)0, acc[1][0], false, false);
    acc[1][1] = __builtin_amdgcn_wmma_f32_16x16x32_bf16(
        false, a1, false, b1, (short)0, acc[1][1], false, false);
  }

  // Epilogue: C/D layout: lane n = l16; VGPR j -> row m = j + khalf*8.
#pragma unroll
  for (int im = 0; im < 2; ++im) {
#pragma unroll
    for (int in = 0; in < 2; ++in) {
      int n = nblk + in * 16 + l16;
      float bias = Ba2[n];
#pragma unroll
      for (int j = 0; j < 8; ++j) {
        int m = mblk + im * 16 + khalf * 8 + j;
        C[(size_t)m * N + n] = (__bf16)(acc[im][in][j] + bias);
      }
    }
  }
}

// ---------------------------------------------------------------------------
// Kernel 3: zero-fill h (determinism across graph replays).
// ---------------------------------------------------------------------------
__global__ void zero_kernel(float* __restrict__ p, int n) {
  int i = blockIdx.x * 256 + threadIdx.x;
  if (i < n) p[i] = 0.f;
}

// ---------------------------------------------------------------------------
// Kernel 4: one recurrent step. grid = B blocks (one per batch row),
// 256 threads = 8 wave32s per block.
// ---------------------------------------------------------------------------
__global__ __launch_bounds__(256) void aru_step_kernel(
    const __bf16* __restrict__ xbf,   // (B,T,F) bf16
    const __bf16* __restrict__ uah,   // (B,T,F) bf16 = x@Ua + Ba2
    const float* __restrict__ Wa,     // (U,F)
    const float* __restrict__ Va,     // (F)
    const float* __restrict__ Ba1,    // (F)
    const float* __restrict__ gk,     // (F,3U)
    const float* __restrict__ grk,    // (U,3U)
    const float* __restrict__ gb,     // (2,3U)
    float* __restrict__ h,            // (B,U) state (read then written)
    float* __restrict__ out,          // (B,T,U)
    int t) {
  __shared__ float was[F_];
  __shared__ float va[F_];
  __shared__ float ctx[F_];
  __shared__ float sc[T_];
  __shared__ float hl[U_];
  __shared__ float xzp[3][G3_];
  __shared__ float xz[G3_];
  __shared__ float hz[G3_];

  const int tid  = threadIdx.x;
  const int lane = tid & 31;
  const int wave = tid >> 5;
  const int b    = blockIdx.x;

  if (tid < U_) hl[tid] = h[b * U_ + tid];
#pragma unroll
  for (int i = 0; i < 4; ++i) va[tid + 256 * i] = Va[tid + 256 * i];
  __syncthreads();

  // ---- Phase A: WaS[f] = Ba1[f] + sum_u h[u]*Wa[u,f] ----
#pragma unroll
  for (int i = 0; i < 4; ++i) {
    int f = tid + 256 * i;
    float acc = Ba1[f];
#pragma unroll
    for (int u = 0; u < U_; ++u) acc = fmaf(hl[u], Wa[u * F_ + f], acc);
    was[f] = acc;
  }
  __syncthreads();

  // ---- Phase B: scores[tau] = sum_f tanh(UaH + WaS) * Va ----
  const __bf16* uahb = uah + (size_t)b * T_ * F_;
  for (int row = wave; row < T_; row += 8) {
    const __bf16* rp = uahb + (size_t)row * F_;
    if (row + 8 < T_)  // stream next row for this wave through the caches
      __builtin_prefetch(rp + 8 * F_ + lane * 8, 0, 3);
    float acc = 0.f;
#pragma unroll
    for (int i = 0; i < 4; ++i) {
      int f = i * 256 + lane * 8;
      bf8v uv = *(const bf8v*)(rp + f);
#pragma unroll
      for (int j = 0; j < 8; ++j)
        acc = fmaf(fast_tanh((float)uv[j] + was[f + j]), va[f + j], acc);
    }
#pragma unroll
    for (int off = 16; off; off >>= 1) acc += __shfl_xor(acc, off, 32);
    if (lane == 0) sc[row] = acc;  // Ba3 omitted: softmax shift-invariant
  }
  __syncthreads();

  // ---- Phase C: softmax over T (wave 0 only) ----
  if (wave == 0) {
    float m = -1e30f;
    for (int j = lane; j < T_; j += 32) m = fmaxf(m, sc[j]);
#pragma unroll
    for (int off = 16; off; off >>= 1) m = fmaxf(m, __shfl_xor(m, off, 32));
    float s = 0.f;
    for (int j = lane; j < T_; j += 32) {
      float e = __expf(sc[j] - m);
      sc[j] = e;
      s += e;
    }
#pragma unroll
    for (int off = 16; off; off >>= 1) s += __shfl_xor(s, off, 32);
    float inv = 1.f / s;
    for (int j = lane; j < T_; j += 32) sc[j] *= inv;
  }
  __syncthreads();

  // ---- Phase D: ctx[f] = sum_tau a[tau] * x[b,tau,f] ----
  {
    const __bf16* xb = xbf + (size_t)b * T_ * F_;
    const int f0 = tid * 4;
    float a0 = 0.f, a1 = 0.f, a2 = 0.f, a3 = 0.f;
    for (int tau = 0; tau < T_; ++tau) {
      float w = sc[tau];
      bf4v xv = *(const bf4v*)(xb + (size_t)tau * F_ + f0);
      a0 = fmaf(w, (float)xv[0], a0);
      a1 = fmaf(w, (float)xv[1], a1);
      a2 = fmaf(w, (float)xv[2], a2);
      a3 = fmaf(w, (float)xv[3], a3);
    }
    ctx[f0 + 0] = a0; ctx[f0 + 1] = a1; ctx[f0 + 2] = a2; ctx[f0 + 3] = a3;
  }
  __syncthreads();

  // ---- Phase E: GRU cell (reset_after=true) ----
  if (tid < 252) {  // 3 partial sums per output column j
    int part = tid / G3_;
    int j = tid - part * G3_;
    int fb = (part == 0) ? 0 : 342 + (part - 1) * 341;
    int fe = fb + ((part == 0) ? 342 : 341);
    float acc = 0.f;
    for (int f = fb; f < fe; ++f) acc = fmaf(ctx[f], gk[f * G3_ + j], acc);
    xzp[part][j] = acc;
  }
  __syncthreads();
  if (tid < G3_) {
    xz[tid] = xzp[0][tid] + xzp[1][tid] + xzp[2][tid] + gb[tid];
    float acc = gb[G3_ + tid];
#pragma unroll
    for (int u = 0; u < U_; ++u) acc = fmaf(hl[u], grk[u * G3_ + tid], acc);
    hz[tid] = acc;
  }
  __syncthreads();
  if (tid < U_) {
    float z  = fast_sigmoid(xz[tid] + hz[tid]);
    float r  = fast_sigmoid(xz[U_ + tid] + hz[U_ + tid]);
    float hh = fast_tanh(xz[2 * U_ + tid] + r * hz[2 * U_ + tid]);
    float hn = z * hl[tid] + (1.f - z) * hh;
    h[b * U_ + tid] = hn;
    out[((size_t)b * T_ + t) * U_ + tid] = hn;
  }
}

// ---------------------------------------------------------------------------
extern "C" void kernel_launch(void* const* d_in, const int* in_sizes, int n_in,
                              void* d_out, int out_size, void* d_ws,
                              size_t ws_size, hipStream_t stream) {
  const float* x   = (const float*)d_in[0];
  const float* Wa  = (const float*)d_in[1];
  const float* Ua  = (const float*)d_in[2];
  const float* Va  = (const float*)d_in[3];
  const float* Ba1 = (const float*)d_in[4];
  const float* Ba2 = (const float*)d_in[5];
  // d_in[6] = Ba3: uniform shift of scores -> cancelled by softmax.
  const float* gk  = (const float*)d_in[7];
  const float* grk = (const float*)d_in[8];
  const float* gb  = (const float*)d_in[9];
  float* out = (float*)d_out;

  char* ws = (char*)d_ws;
  const size_t SZ_XBF = (size_t)B_ * T_ * F_ * 2;  // 78,643,200
  const size_t SZ_UAT = (size_t)F_ * F_ * 2;       //  2,097,152
  const size_t SZ_UAH = SZ_XBF;                    // 78,643,200
  __bf16* xbf = (__bf16*)ws;
  __bf16* uat = (__bf16*)(ws + SZ_XBF);
  __bf16* uah = (__bf16*)(ws + SZ_XBF + SZ_UAT);
  float*  h   = (float*)(ws + SZ_XBF + SZ_UAT + SZ_UAH);

  convert_kernel<<<4096, 256, 0, stream>>>(x, Ua, xbf, uat);
  wmma_gemm_kernel<<<dim3(F_ / 64, (B_ * T_) / 128), 256, 0, stream>>>(
      xbf, uat, Ba2, uah);
  zero_kernel<<<(B_ * U_ + 255) / 256, 256, 0, stream>>>(h, B_ * U_);
  for (int t = 0; t < T_; ++t) {
    aru_step_kernel<<<B_, 256, 0, stream>>>(xbf, uah, Wa, Va, Ba1, gk, grk,
                                            gb, h, out, t);
  }
}
